// DEQDotProductAttentionTransformerMD17_16544214024827
// MI455X (gfx1250) — matrix operand order repaired and checked by reading, not compiled, not run on hardware
//
// MI455X (gfx1250) implementation of the DEQ equivariant attention transformer.
// All dense contractions lowered to v_wmma_f32_16x16x32_f16 (wave32 WMMA),
// scatter/softmax/layernorm as bandwidth-bound VALU kernels with atomics.
// GEMM: 64x64 block tile, 8 waves, 2 WMMA accumulators per wave (B-frag reuse),
// LDS A-tile padded to 17 dwords/row to kill 4-way bank conflicts.
#include <hip/hip_runtime.h>
#include <hip/hip_bf16.h>
#include <stdint.h>

typedef __attribute__((ext_vector_type(16))) _Float16 v16h;
typedef __attribute__((ext_vector_type(8)))  float    v8f;

static constexpr int kN = 10000, kE = 50000, kG = 500;
static constexpr int kC0 = 128, kC1 = 64, kC2 = 32, kD = 480;
static constexpr int kNB = 128, kNW = 320, kFD = 512;

union Frag16 { v16h v; unsigned u[8]; };

__device__ __forceinline__ unsigned pack2h(float f0, float f1) {
  unsigned h0 = (unsigned)__builtin_bit_cast(unsigned short, (_Float16)f0);
  unsigned h1 = (unsigned)__builtin_bit_cast(unsigned short, (_Float16)f1);
  return h0 | (h1 << 16);
}

// ---------------------------------------------------------------------------
// Generic strided GEMM, C[M,Np] = act(A[M,K] @ B[K,Np]) (+ R), via WMMA f16.
// A element (r,k) at A[r*lda + k*sA]; B row-major (ldb); C at C[r*ldc + n*sC].
// act: 0=none 1=silu 2=sigmoid. K must be a multiple of 32 (true for all uses).
// Block: 256 threads = 8 waves (2x4); block tile 64x64; each wave owns two
// 16x16 output tiles stacked in M, sharing one B fragment per k-step.
// ---------------------------------------------------------------------------
__global__ __launch_bounds__(256) void gemm_f16_wmma(
    const float* __restrict__ A, long lda, int sA,
    const float* __restrict__ B, int ldb,
    const float* __restrict__ R, long ldr, int sR,
    float* __restrict__ C, long ldc, int sC,
    int M, int Np, int K, int act)
{
  __shared__ unsigned la[64][17];   // [row][k-pair], padded row (bank-conflict free)
  __shared__ unsigned lb[16][64];   // [k-pair][col]
  const int tid  = threadIdx.x;
  const int wave = tid >> 5, lane = tid & 31;
  const int wm = wave & 1, wn = wave >> 1;           // 2 row-groups x 4 col-tiles
  const int br = blockIdx.y * 64, bc = blockIdx.x * 64;
  v8f acc0 = {}, acc1 = {};
  for (int k0 = 0; k0 < K; k0 += 32) {
    // Stage A tile (64x32 f32 -> f16 pairs), 1024 dwords / 256 threads.
#pragma unroll
    for (int i = tid; i < 1024; i += 256) {
      int r = i >> 4, pp = i & 15;
      int gr = br + r;
      float f0 = 0.f, f1 = 0.f;
      if (gr < M) {
        const float* ap = A + (long)gr * lda + (long)(k0 + 2 * pp) * sA;
        f0 = ap[0]; f1 = ap[sA];
        if (pp == 0 && k0 + 32 < K)
          __builtin_prefetch(ap + (long)32 * sA, 0, 0);  // global_prefetch_b8
      }
      la[r][pp] = pack2h(f0, f1);
    }
    // Stage B tile (32x64), 1024 dwords / 256 threads.
#pragma unroll
    for (int i = tid; i < 1024; i += 256) {
      int pp = i >> 6, nn = i & 63;
      int gk = k0 + 2 * pp, gn = bc + nn;
      float f0 = 0.f, f1 = 0.f;
      if (gn < Np) {
        f0 = B[(long)gk * ldb + gn];
        f1 = B[(long)(gk + 1) * ldb + gn];
      }
      lb[pp][nn] = pack2h(f0, f1);
    }
    __syncthreads();
    // Gather fragments per the gfx1250 16-bit A(16x32)/B(32x16) layouts.
    Frag16 a0, a1, b;
    const int  hl = lane & 15;
    const bool hi = lane >= 16;
    const int  ar = wm * 32 + hl;
    const int  bn = wn * 16 + hl;
#pragma unroll
    for (int j = 0; j < 8; ++j) {
      int pa = (j < 4 ? j : j + 4) + (hi ? 4 : 0);  // A: K 0..7,16..23 | 8..15,24..31
      a0.u[j] = la[ar][pa];
      a1.u[j] = la[ar + 16][pa];
      b.u[j]  = lb[pa][bn];                         // B: K 0..7,16..23 | 8..15,24..31
    }
    acc0 = __builtin_amdgcn_wmma_f32_16x16x32_f16(
        false, a0.v, false, b.v, (short)0, acc0, false, false);
    acc1 = __builtin_amdgcn_wmma_f32_16x16x32_f16(
        false, a1.v, false, b.v, (short)0, acc1, false, false);
    __syncthreads();
  }
  const int col = bc + wn * 16 + (lane & 15);
  if (col >= Np) return;
  const long rcol = (long)col;
  const int rb = br + wm * 32 + ((lane < 16) ? 0 : 8);
#pragma unroll
  for (int half = 0; half < 2; ++half) {
    const v8f& acc = half ? acc1 : acc0;
#pragma unroll
    for (int r = 0; r < 8; ++r) {
      int row = rb + half * 16 + r;
      if (row >= M) continue;
      float v = acc[r];
      if (act == 1)      v = v / (1.f + __expf(-v));   // silu
      else if (act == 2) v = 1.f / (1.f + __expf(-v)); // sigmoid
      if (R) v += R[(long)row * ldr + rcol * sR];
      C[(long)row * ldc + rcol * sC] = v;
    }
  }
}

// ---------------------------------------------------------------------------
// Elementwise / graph kernels
// ---------------------------------------------------------------------------
__global__ void fill_f32(float* p, float v, long n) {
  long i = (long)blockIdx.x * blockDim.x + threadIdx.x;
  if (i < n) p[i] = v;
}
__global__ void fill_u32(unsigned* p, unsigned v, long n) {
  long i = (long)blockIdx.x * blockDim.x + threadIdx.x;
  if (i < n) p[i] = v;
}

__global__ __launch_bounds__(128) void edge_geom_kernel(
    const float* __restrict__ pos, const int* __restrict__ src,
    const int* __restrict__ dst, float* __restrict__ rbf,
    float* __restrict__ y1, float* __restrict__ y2)
{
  int e = blockIdx.x, t = threadIdx.x;
  int s = src[e], d = dst[e];
  float vx = pos[d*3+0] - pos[s*3+0];
  float vy = pos[d*3+1] - pos[s*3+1];
  float vz = pos[d*3+2] - pos[s*3+2];
  float r = sqrtf(vx*vx + vy*vy + vz*vz + 1e-12f);
  float x = vx / r, y = vy / r, z = vz / r;
  float mu = 5.0f * (float)t / 127.0f;
  float sg = 5.0f / 128.0f;
  float dd = (r - mu) / sg;
  rbf[(long)e * kNB + t] = __expf(-0.5f * dd * dd);
  if (t == 0) {
    const float s3 = 1.7320508075688772f, s15 = 3.872983346207417f;
    y1[e*3+0] = s3 * x; y1[e*3+1] = s3 * y; y1[e*3+2] = s3 * z;
    y2[e*5+0] = s15 * x * y;
    y2[e*5+1] = s15 * y * z;
    y2[e*5+2] = 1.1180339887498949f * (3.f * z * z - 1.f);
    y2[e*5+3] = s15 * x * z;
    y2[e*5+4] = 1.9364916731037085f * (x * x - y * y);
  }
}

__global__ void gather_embed_kernel(const int* __restrict__ z,
                                    const float* __restrict__ tab,
                                    float* __restrict__ s_atom)
{
  long i = (long)blockIdx.x * blockDim.x + threadIdx.x;
  if (i >= (long)kN * kC0) return;
  int n = (int)(i / kC0), c = (int)(i % kC0);
  s_atom[i] = tab[(long)z[n] * kC0 + c];
}

// Degree-embedding scatter: modulate with v_e = 0 then segment_sum over dst.
__global__ __launch_bounds__(512) void deg_scatter_kernel(
    const float* __restrict__ s_atom, const float* __restrict__ p1n,
    const float* __restrict__ p2n, const float* __restrict__ wd /*E x 320*/,
    const float* __restrict__ y1, const float* __restrict__ y2,
    const int* __restrict__ src, const int* __restrict__ dst,
    float* __restrict__ deg)
{
  int e = blockIdx.x, d = threadIdx.x;
  if (d >= kD) return;
  int s = src[e], t = dst[e];
  const float* w = wd + (long)e * kNW;
  float val;
  if (d < 128) {
    val = s_atom[(long)s * kC0 + d] * w[d];
  } else if (d < 320) {
    int c = (d - 128) / 3, m = (d - 128) % 3;
    val = p1n[(long)s * kC1 + c] * y1[e*3+m] * w[192 + c];
  } else {
    int c = (d - 320) / 5, m = (d - 320) % 5;
    val = p2n[(long)s * kC2 + c] * y2[e*5+m] * w[288 + c];
  }
  atomicAdd(&deg[(long)t * kD + d], val);
}

__global__ void inj_kernel(const float* __restrict__ deg,
                           const float* __restrict__ s_atom,
                           float* __restrict__ inj)
{
  long i = (long)blockIdx.x * blockDim.x + threadIdx.x;
  if (i >= (long)kN * kD) return;
  int d = (int)(i % kD);
  float v = deg[i] * 0.25f;               // 1/sqrt(AVG_DEGREE)
  if (d < kC0) v += s_atom[(i / kD) * kC0 + d];
  inj[i] = v;
}

__global__ void add_kernel(float* __restrict__ x, const float* __restrict__ a, long n) {
  long i = (long)blockIdx.x * blockDim.x + threadIdx.x;
  if (i < n) x[i] += a[i];
}

// Equivariant layernorm over one node per block (128 threads).
__global__ __launch_bounds__(128) void eqln_kernel(const float* __restrict__ x,
                                                   float* __restrict__ xn)
{
  __shared__ float red[128];
  int n = blockIdx.x, t = threadIdx.x;
  const float* xr = x  + (long)n * kD;
  float*       xo = xn + (long)n * kD;
  float s = xr[t];
  red[t] = s; __syncthreads();
  for (int o = 64; o > 0; o >>= 1) { if (t < o) red[t] += red[t + o]; __syncthreads(); }
  float mean = red[0] * (1.f / 128.f); __syncthreads();
  float c = s - mean;
  red[t] = c * c; __syncthreads();
  for (int o = 64; o > 0; o >>= 1) { if (t < o) red[t] += red[t + o]; __syncthreads(); }
  xo[t] = c * rsqrtf(red[0] * (1.f / 128.f) + 1e-5f);
  __syncthreads();
  float acc = 0.f;
  for (int d = t; d < 192; d += 128) { float v = xr[128 + d]; acc += v * v; }
  red[t] = acc; __syncthreads();
  for (int o = 64; o > 0; o >>= 1) { if (t < o) red[t] += red[t + o]; __syncthreads(); }
  float sc1 = rsqrtf(red[0] * (1.f / 64.f) + 1e-5f); __syncthreads();
  for (int d = t; d < 192; d += 128) xo[128 + d] = xr[128 + d] * sc1;
  acc = 0.f;
  for (int d = t; d < 160; d += 128) { float v = xr[320 + d]; acc += v * v; }
  red[t] = acc; __syncthreads();
  for (int o = 64; o > 0; o >>= 1) { if (t < o) red[t] += red[t + o]; __syncthreads(); }
  float sc2 = rsqrtf(red[0] * (1.f / 32.f) + 1e-5f); __syncthreads();
  for (int d = t; d < 160; d += 128) xo[320 + d] = xr[320 + d] * sc2;
}

__device__ __forceinline__ int head_of(int d) {
  if (d < 128) return d >> 5;
  if (d < 320) return ((d - 128) / 3) >> 4;
  return ((d - 320) / 5) >> 3;
}

// Per-edge: k,v via modulation; v stored; logits[e,h] = q[dst].k / sqrt(HD).
__global__ __launch_bounds__(512) void attn_edge_kernel(
    const float* __restrict__ xn, const float* __restrict__ q,
    const float* __restrict__ p1n, const float* __restrict__ p2n,
    const float* __restrict__ w /*E x 640*/, const float* __restrict__ y1,
    const float* __restrict__ y2, const int* __restrict__ src,
    const int* __restrict__ dst, float* __restrict__ ve,
    float* __restrict__ lg)
{
  __shared__ float hl[4];
  int e = blockIdx.x, d = threadIdx.x;
  if (d < 4) hl[d] = 0.f;
  __syncthreads();
  if (d < kD) {
    int s = src[e], t = dst[e];
    const float* wk = w + (long)e * 640;
    const float* wv = wk + kNW;
    float kd, vd; int h;
    float xe = xn[(long)s * kD + d];
    if (d < 128) {
      kd = xe * wk[d]; vd = xe * wv[d]; h = d >> 5;
    } else if (d < 320) {
      int c = (d - 128) / 3, m = (d - 128) % 3;
      float pr = p1n[(long)s * kC1 + c] * y1[e*3+m];
      kd = xe * wk[128 + c] + pr * wk[192 + c];
      vd = xe * wv[128 + c] + pr * wv[192 + c];
      h = c >> 4;
    } else {
      int c = (d - 320) / 5, m = (d - 320) % 5;
      float pr = p2n[(long)s * kC2 + c] * y2[e*5+m];
      kd = xe * wk[256 + c] + pr * wk[288 + c];
      vd = xe * wv[256 + c] + pr * wv[288 + c];
      h = c >> 3;
    }
    ve[(long)e * kD + d] = vd;
    atomicAdd(&hl[h], q[(long)t * kD + d] * kd);   // ds_add_f32
  }
  __syncthreads();
  if (d < 4) lg[(long)e * 4 + d] = hl[d] * 0.09128709291752768f; // 1/sqrt(120)
}

__device__ __forceinline__ unsigned enc_key(float f) {
  unsigned b = __float_as_uint(f);
  return (b & 0x80000000u) ? ~b : (b | 0x80000000u);
}
__device__ __forceinline__ float dec_key(unsigned k) {
  return __uint_as_float((k & 0x80000000u) ? (k ^ 0x80000000u) : ~k);
}

__global__ void segmax_kernel(const float* __restrict__ lg,
                              const int* __restrict__ dst,
                              unsigned* __restrict__ mb)
{
  long i = (long)blockIdx.x * blockDim.x + threadIdx.x;
  if (i >= (long)kE * 4) return;
  int e = (int)(i >> 2), h = (int)(i & 3);
  atomicMax(&mb[(long)dst[e] * 4 + h], enc_key(lg[i]));
}

__global__ void expden_kernel(float* __restrict__ lg,
                              const unsigned* __restrict__ mb,
                              const int* __restrict__ dst,
                              float* __restrict__ den)
{
  long i = (long)blockIdx.x * blockDim.x + threadIdx.x;
  if (i >= (long)kE * 4) return;
  int e = (int)(i >> 2), h = (int)(i & 3);
  int t = dst[e];
  float ex = __expf(lg[i] - dec_key(mb[(long)t * 4 + h]));
  lg[i] = ex;
  atomicAdd(&den[(long)t * 4 + h], ex);
}

__global__ __launch_bounds__(512) void scatter_att_kernel(
    const float* __restrict__ lg, const float* __restrict__ den,
    const float* __restrict__ ve, const int* __restrict__ dst,
    float* __restrict__ att)
{
  int e = blockIdx.x, d = threadIdx.x;
  if (d >= kD) return;
  int t = dst[e], h = head_of(d);
  float alpha = lg[(long)e * 4 + h] / (den[(long)t * 4 + h] + 1e-9f);
  atomicAdd(&att[(long)t * kD + d], alpha * ve[(long)e * kD + d]);
}

__global__ void gate_kernel(float* __restrict__ y, const float* __restrict__ g) {
  long i = (long)blockIdx.x * blockDim.x + threadIdx.x;
  if (i >= (long)kN * kD) return;
  long n = i / kD; int d = (int)(i % kD);
  float v = y[i];
  if (d < 128)      y[i] = v / (1.f + __expf(-v));
  else if (d < 320) y[i] = v * g[n * 96 + (d - 128) / 3];
  else              y[i] = v * g[n * 96 + 64 + (d - 320) / 5];
}

__global__ __launch_bounds__(256) void featln_kernel(float* __restrict__ feat) {
  __shared__ float red[256];
  int n = blockIdx.x, t = threadIdx.x;
  float* f = feat + (long)n * kFD;
  float a = f[t], b = f[t + 256];
  red[t] = a + b; __syncthreads();
  for (int o = 128; o > 0; o >>= 1) { if (t < o) red[t] += red[t + o]; __syncthreads(); }
  float mean = red[0] * (1.f / 512.f); __syncthreads();
  float ca = a - mean, cb = b - mean;
  red[t] = ca * ca + cb * cb; __syncthreads();
  for (int o = 128; o > 0; o >>= 1) { if (t < o) red[t] += red[t + o]; __syncthreads(); }
  float sc = rsqrtf(red[0] * (1.f / 512.f) + 1e-5f);
  f[t] = ca * sc; f[t + 256] = cb * sc;
}

__global__ __launch_bounds__(256) void final_kernel(
    const float* __restrict__ h, const float* __restrict__ Wh2,
    const int* __restrict__ bseg, float* __restrict__ out)
{
  __shared__ float red[256];
  int n = blockIdx.x, t = threadIdx.x;
  const float* hr = h + (long)n * kFD;
  red[t] = hr[t] * Wh2[t] + hr[t + 256] * Wh2[t + 256];
  __syncthreads();
  for (int o = 128; o > 0; o >>= 1) { if (t < o) red[t] += red[t + o]; __syncthreads(); }
  if (t == 0) atomicAdd(&out[bseg[n]], red[0] * 0.2182178902359924f); // 1/sqrt(21)
}

// ---------------------------------------------------------------------------
// Host orchestration
// ---------------------------------------------------------------------------
static void gemm(hipStream_t st, const float* A, long lda, int sA,
                 const float* B, int ldb, const float* R, long ldr, int sR,
                 float* C, long ldc, int sC, int M, int Np, int K, int act)
{
  dim3 g((Np + 63) / 64, (M + 63) / 64);
  gemm_f16_wmma<<<g, 256, 0, st>>>(A, lda, sA, B, ldb, R, ldr, sR,
                                   C, ldc, sC, M, Np, K, act);
}

struct Ctx {
  const float *rbf, *y1, *y2;
  const int *esrc, *edst;
  const float *Wq0, *Wq1, *Wq2, *P1, *P2, *R1, *R2, *R3;
  const float *Wp0, *Wp1, *Wp2, *Wg, *Wi0, *Wi1, *Wi2, *Wo0, *Wo1, *Wo2;
  float *x, *xn, *q, *p1n, *p2n, *h1, *h2, *w, *ve, *lg, *den, *att, *ybuf, *g;
  unsigned* mb;
};

static void run_block(hipStream_t st, const Ctx& c, int i) {
  const float* Wq0i = c.Wq0 + (long)i * 128 * 128;
  const float* Wq1i = c.Wq1 + (long)i * 64 * 64;
  const float* Wq2i = c.Wq2 + (long)i * 32 * 32;
  const float* P1i  = c.P1  + (long)i * 128 * 64;
  const float* P2i  = c.P2  + (long)i * 128 * 32;
  const float* R1i  = c.R1  + (long)i * 128 * 64;
  const float* R2i  = c.R2  + (long)i * 64 * 64;
  const float* R3i  = c.R3  + (long)i * 64 * 640;
  const float* Wp0i = c.Wp0 + (long)i * 128 * 128;
  const float* Wp1i = c.Wp1 + (long)i * 64 * 64;
  const float* Wp2i = c.Wp2 + (long)i * 32 * 32;
  const float* Wgi  = c.Wg  + (long)i * 128 * 96;
  const float* Wi0i = c.Wi0 + (long)i * 128 * 128;
  const float* Wi1i = c.Wi1 + (long)i * 64 * 64;
  const float* Wi2i = c.Wi2 + (long)i * 32 * 32;
  const float* Wo0i = c.Wo0 + (long)i * 128 * 128;
  const float* Wo1i = c.Wo1 + (long)i * 64 * 64;
  const float* Wo2i = c.Wo2 + (long)i * 32 * 32;

  eqln_kernel<<<kN, 128, 0, st>>>(c.x, c.xn);
  // q = eq_linear(xn, Wq)
  gemm(st, c.xn, kD, 1, Wq0i, 128, nullptr, 0, 0, c.q, kD, 1, kN, 128, 128, 0);
  for (int m = 0; m < 3; ++m)
    gemm(st, c.xn + 128 + m, kD, 3, Wq1i, 64, nullptr, 0, 0, c.q + 128 + m, kD, 3, kN, 64, 64, 0);
  for (int m = 0; m < 5; ++m)
    gemm(st, c.xn + 320 + m, kD, 5, Wq2i, 32, nullptr, 0, 0, c.q + 320 + m, kD, 5, kN, 32, 32, 0);
  // radial MLP -> w (E x 640)
  gemm(st, c.rbf, 128, 1, R1i, 64, nullptr, 0, 0, c.h1, 64, 1, kE, 64, 128, 1);
  gemm(st, c.h1, 64, 1, R2i, 64, nullptr, 0, 0, c.h2, 64, 1, kE, 64, 64, 1);
  gemm(st, c.h2, 64, 1, R3i, 640, nullptr, 0, 0, c.w, 640, 1, kE, 640, 64, 0);
  // node projections for modulation
  gemm(st, c.xn, kD, 1, P1i, 64, nullptr, 0, 0, c.p1n, 64, 1, kN, 64, 128, 0);
  gemm(st, c.xn, kD, 1, P2i, 32, nullptr, 0, 0, c.p2n, 32, 1, kN, 32, 128, 0);
  // attention: logits, segment softmax, weighted scatter
  attn_edge_kernel<<<kE, 512, 0, st>>>(c.xn, c.q, c.p1n, c.p2n, c.w, c.y1, c.y2,
                                       c.esrc, c.edst, c.ve, c.lg);
  fill_u32<<<(kN * 4 + 255) / 256, 256, 0, st>>>(c.mb, 0x007fffffu, (long)kN * 4);
  segmax_kernel<<<(kE * 4 + 255) / 256, 256, 0, st>>>(c.lg, c.edst, c.mb);
  fill_f32<<<(kN * 4 + 255) / 256, 256, 0, st>>>(c.den, 0.f, (long)kN * 4);
  expden_kernel<<<(kE * 4 + 255) / 256, 256, 0, st>>>(c.lg, c.mb, c.edst, c.den);
  fill_f32<<<((long)kN * kD + 255) / 256, 256, 0, st>>>(c.att, 0.f, (long)kN * kD);
  scatter_att_kernel<<<kE, 512, 0, st>>>(c.lg, c.den, c.ve, c.edst, c.att);
  // x += eq_linear(att, Wp)   (residual fused into GEMM epilogue)
  gemm(st, c.att, kD, 1, Wp0i, 128, c.x, kD, 1, c.x, kD, 1, kN, 128, 128, 0);
  for (int m = 0; m < 3; ++m)
    gemm(st, c.att + 128 + m, kD, 3, Wp1i, 64, c.x + 128 + m, kD, 3, c.x + 128 + m, kD, 3, kN, 64, 64, 0);
  for (int m = 0; m < 5; ++m)
    gemm(st, c.att + 320 + m, kD, 5, Wp2i, 32, c.x + 320 + m, kD, 5, c.x + 320 + m, kD, 5, kN, 32, 32, 0);
  // gated MLP half
  eqln_kernel<<<kN, 128, 0, st>>>(c.x, c.xn);
  gemm(st, c.xn, kD, 1, Wi0i, 128, nullptr, 0, 0, c.ybuf, kD, 1, kN, 128, 128, 0);
  for (int m = 0; m < 3; ++m)
    gemm(st, c.xn + 128 + m, kD, 3, Wi1i, 64, nullptr, 0, 0, c.ybuf + 128 + m, kD, 3, kN, 64, 64, 0);
  for (int m = 0; m < 5; ++m)
    gemm(st, c.xn + 320 + m, kD, 5, Wi2i, 32, nullptr, 0, 0, c.ybuf + 320 + m, kD, 5, kN, 32, 32, 0);
  gemm(st, c.xn, kD, 1, Wgi, 96, nullptr, 0, 0, c.g, 96, 1, kN, 96, 128, 2); // sigmoid
  gate_kernel<<<((long)kN * kD + 255) / 256, 256, 0, st>>>(c.ybuf, c.g);
  // x += eq_linear(y, Wo)
  gemm(st, c.ybuf, kD, 1, Wo0i, 128, c.x, kD, 1, c.x, kD, 1, kN, 128, 128, 0);
  for (int m = 0; m < 3; ++m)
    gemm(st, c.ybuf + 128 + m, kD, 3, Wo1i, 64, c.x + 128 + m, kD, 3, c.x + 128 + m, kD, 3, kN, 64, 64, 0);
  for (int m = 0; m < 5; ++m)
    gemm(st, c.ybuf + 320 + m, kD, 5, Wo2i, 32, c.x + 320 + m, kD, 5, c.x + 320 + m, kD, 5, kN, 32, 32, 0);
}

extern "C" void kernel_launch(void* const* d_in, const int* in_sizes, int n_in,
                              void* d_out, int out_size, void* d_ws, size_t ws_size,
                              hipStream_t stream) {
  const float* pos  = (const float*)d_in[0];
  const int* atom_z = (const int*)d_in[1];
  const int* esrc   = (const int*)d_in[2];
  const int* edst   = (const int*)d_in[3];
  const int* bseg   = (const int*)d_in[4];
  const float* atab = (const float*)d_in[5];
  const float* Pd1  = (const float*)d_in[6];
  const float* Pd2  = (const float*)d_in[7];
  const float* Rd1  = (const float*)d_in[8];
  const float* Rd2  = (const float*)d_in[9];
  const float* Rd3  = (const float*)d_in[10];
  Ctx c;
  c.Wq0 = (const float*)d_in[11]; c.Wq1 = (const float*)d_in[12];
  c.Wq2 = (const float*)d_in[13]; c.P1  = (const float*)d_in[14];
  c.P2  = (const float*)d_in[15]; c.R1  = (const float*)d_in[16];
  c.R2  = (const float*)d_in[17]; c.R3  = (const float*)d_in[18];
  c.Wp0 = (const float*)d_in[19]; c.Wp1 = (const float*)d_in[20];
  c.Wp2 = (const float*)d_in[21]; c.Wg  = (const float*)d_in[22];
  c.Wi0 = (const float*)d_in[23]; c.Wi1 = (const float*)d_in[24];
  c.Wi2 = (const float*)d_in[25]; c.Wo0 = (const float*)d_in[26];
  c.Wo1 = (const float*)d_in[27]; c.Wo2 = (const float*)d_in[28];
  const float* Wf  = (const float*)d_in[29];
  const float* Wh1 = (const float*)d_in[30];
  const float* Wh2 = (const float*)d_in[31];
  float* out = (float*)d_out;
  c.esrc = esrc; c.edst = edst;

  // Carve workspace (floats).
  float* p = (float*)d_ws;
  auto alloc = [&](long nfl) { float* r = p; p += nfl; return r; };
  float* rbf   = alloc((long)kE * kNB);
  float* y1    = alloc((long)kE * 3);
  float* y2    = alloc((long)kE * 5);
  float* satom = alloc((long)kN * kC0);
  float* pd1n  = alloc((long)kN * kC1);
  float* pd2n  = alloc((long)kN * kC2);
  c.p1n  = alloc((long)kN * kC1);
  c.p2n  = alloc((long)kN * kC2);
  c.h1   = alloc((long)kE * 64);
  c.h2   = alloc((long)kE * 64);
  c.w    = alloc((long)kE * 640);
  c.x    = alloc((long)kN * kD);
  float* inj = alloc((long)kN * kD);
  c.xn   = alloc((long)kN * kD);
  c.q    = alloc((long)kN * kD);
  c.ybuf = alloc((long)kN * kD);
  c.att  = alloc((long)kN * kD);
  c.g    = alloc((long)kN * 96);
  c.ve   = alloc((long)kE * kD);
  c.lg   = alloc((long)kE * 4);
  c.den  = alloc((long)kN * 4);
  c.mb   = (unsigned*)alloc((long)kN * 4);
  float* feat = alloc((long)kN * kFD);
  float* hfin = alloc((long)kN * kFD);
  c.rbf = rbf; c.y1 = y1; c.y2 = y2;

  // --- precompute: geometry, embedding, degree injection ---
  edge_geom_kernel<<<kE, 128, 0, stream>>>(pos, esrc, edst, rbf, y1, y2);
  gather_embed_kernel<<<((long)kN * kC0 + 255) / 256, 256, 0, stream>>>(atom_z, atab, satom);
  gemm(stream, satom, 128, 1, Pd1, 64, nullptr, 0, 0, pd1n, 64, 1, kN, 64, 128, 0);
  gemm(stream, satom, 128, 1, Pd2, 32, nullptr, 0, 0, pd2n, 32, 1, kN, 32, 128, 0);
  gemm(stream, rbf, 128, 1, Rd1, 64, nullptr, 0, 0, c.h1, 64, 1, kE, 64, 128, 1);
  gemm(stream, c.h1, 64, 1, Rd2, 64, nullptr, 0, 0, c.h2, 64, 1, kE, 64, 64, 1);
  gemm(stream, c.h2, 64, 1, Rd3, 320, nullptr, 0, 0, c.w, 320, 1, kE, 320, 64, 0);
  fill_f32<<<((long)kN * kD + 255) / 256, 256, 0, stream>>>(c.att, 0.f, (long)kN * kD);
  deg_scatter_kernel<<<kE, 512, 0, stream>>>(satom, pd1n, pd2n, c.w, y1, y2, esrc, edst, c.att);
  inj_kernel<<<((long)kN * kD + 255) / 256, 256, 0, stream>>>(c.att, satom, inj);
  fill_f32<<<((long)kN * kD + 255) / 256, 256, 0, stream>>>(c.x, 0.f, (long)kN * kD);

  // --- DEQ iterations: z=0; repeat{ x = z + inj; 5 blocks; z = x } x2; block 5 ---
  for (int t = 0; t < 2; ++t) {
    add_kernel<<<((long)kN * kD + 255) / 256, 256, 0, stream>>>(c.x, inj, (long)kN * kD);
    for (int i = 0; i < 5; ++i) run_block(stream, c, i);
  }
  run_block(stream, c, 5);

  // --- readout head ---
  gemm(stream, c.x, kD, 1, Wf, 512, nullptr, 0, 0, feat, 512, 1, kN, 512, 128, 0);
  featln_kernel<<<kN, 256, 0, stream>>>(feat);
  gemm(stream, feat, 512, 1, Wh1, 512, nullptr, 0, 0, hfin, 512, 1, kN, 512, 512, 1);
  fill_f32<<<(kG + 255) / 256, 256, 0, stream>>>(out, 0.f, (long)kG);
  final_kernel<<<kN, 256, 0, stream>>>(hfin, Wh2, bseg, out);
}
